// Attention_66941360275664
// MI455X (gfx1250) — matrix operand order repaired and checked by reading, not compiled, not run on hardware
//
#include <hip/hip_runtime.h>

// ---------------------------------------------------------------------------
// Types for CDNA5 WMMA (wave32):  D = A(16x32 bf16) * B(32x16 bf16) + C(f32)
// ---------------------------------------------------------------------------
typedef __attribute__((ext_vector_type(16))) __bf16        bf16x16;
typedef __attribute__((ext_vector_type(8)))  float         f32x8;
typedef __attribute__((ext_vector_type(4)))  float         f32x4;
typedef __attribute__((ext_vector_type(4)))  unsigned int  u32x4;

union FragU { bf16x16 v; u32x4 q[2]; };

__device__ __forceinline__ unsigned short f2bf(float f) {
  union { float f; unsigned int u; } x; x.f = f;
  unsigned int r = x.u + 0x7fffu + ((x.u >> 16) & 1u);   // RNE
  return (unsigned short)(r >> 16);
}

__device__ __forceinline__ f32x8 wmma_bf16(const bf16x16& a, const bf16x16& b,
                                           const f32x8& c) {
  return __builtin_amdgcn_wmma_f32_16x16x32_bf16(
      /*neg_a=*/false, a, /*neg_b=*/false, b,
      /*c_mod=*/(short)0, c, /*reuse_a=*/false, /*reuse_b=*/false);
}

__device__ __forceinline__ f32x8 zero8() {
  f32x8 z;
#pragma unroll
  for (int i = 0; i < 8; ++i) z[i] = 0.f;
  return z;
}

// xor-shuffle reductions over each 16-lane half (rows of a 16x16 C tile live
// in lanes 0-15 / 16-31; offsets 1,2,4,8 never cross halves in wave32)
__device__ __forceinline__ float halfmax(float v) {
#pragma unroll
  for (int off = 1; off < 16; off <<= 1) v = fmaxf(v, __shfl_xor(v, off, 32));
  return v;
}
__device__ __forceinline__ float halfsum(float v) {
#pragma unroll
  for (int off = 1; off < 16; off <<= 1) v += __shfl_xor(v, off, 32);
  return v;
}

// Problem dims (fixed by reference)
#define DD   1024
#define SS   2048
#define HH   16
#define HDD  64
#define BB   2
#define MM   (BB * SS)     // 4096 rows
#define N3   (3 * DD)      // 3072

// ---------------------------------------------------------------------------
// Kernel 1: QKV projection.  x[4096,1024] @ w_attn[1024,3072] + b_attn.
// 128x128 tile / block, 256 thr (8 waves), each wave 32x64 = 2x4 WMMA tiles.
// Epilogue: Q,K -> [B,H,S,HD] bf16 ;  V -> transposed [B,H,HD,S] bf16.
// ---------------------------------------------------------------------------
__global__ __launch_bounds__(256) void qkv_kernel(
    const float* __restrict__ X, const float* __restrict__ W,
    const float* __restrict__ bias,
    unsigned short* __restrict__ Qo, unsigned short* __restrict__ Ko,
    unsigned short* __restrict__ Vt) {
  __shared__ __align__(16) unsigned short As[128 * 48];  // [row][k] stride 48
  __shared__ __align__(16) unsigned short Bs[128 * 48];  // [n][k]  stride 48

  const int tid = threadIdx.x;
  const int lane = tid & 31, wave = tid >> 5;
  const int wm = wave & 3, wn = wave >> 2;
  const int g = lane >> 4, l = lane & 15;
  const int m0 = blockIdx.y * 128, n0 = blockIdx.x * 128;

  f32x8 acc[2][4];
#pragma unroll
  for (int mt = 0; mt < 2; ++mt)
#pragma unroll
    for (int nt = 0; nt < 4; ++nt) acc[mt][nt] = zero8();

  for (int k0 = 0; k0 < DD; k0 += 32) {
    // --- stage A tile 128x32 (fp32 -> bf16) ---
#pragma unroll
    for (int i = 0; i < 4; ++i) {
      int slot = tid + i * 256;            // 1024 float4 slots
      int r = slot >> 3, c4 = slot & 7;
      f32x4 a = *(const f32x4*)(X + (size_t)(m0 + r) * DD + k0 + c4 * 4);
      unsigned int lo = (unsigned)f2bf(a[0]) | ((unsigned)f2bf(a[1]) << 16);
      unsigned int hi = (unsigned)f2bf(a[2]) | ((unsigned)f2bf(a[3]) << 16);
      unsigned int* p = (unsigned int*)&As[r * 48 + c4 * 4];
      p[0] = lo; p[1] = hi;
    }
    // --- stage B tile 32x128 transposed into [n][k] (fp32 -> bf16) ---
#pragma unroll
    for (int i = 0; i < 4; ++i) {
      int slot = tid + i * 256;            // 1024 float4 slots
      int kr = slot >> 5, c4 = slot & 31;
      f32x4 w = *(const f32x4*)(W + (size_t)(k0 + kr) * N3 + n0 + c4 * 4);
#pragma unroll
      for (int j = 0; j < 4; ++j) Bs[(c4 * 4 + j) * 48 + kr] = f2bf(w[j]);
    }
    __syncthreads();

    FragU af[2], bfr[4];
#pragma unroll
    for (int mt = 0; mt < 2; ++mt) {
      int row = wm * 32 + mt * 16 + l;
      af[mt].q[0] = *(const u32x4*)(&As[row * 48 + g * 8]);
      af[mt].q[1] = *(const u32x4*)(&As[row * 48 + 16 + g * 8]);
    }
#pragma unroll
    for (int nt = 0; nt < 4; ++nt) {
      int col = wn * 64 + nt * 16 + l;
      bfr[nt].q[0] = *(const u32x4*)(&Bs[col * 48 + g * 8]);
      bfr[nt].q[1] = *(const u32x4*)(&Bs[col * 48 + 16 + g * 8]);
    }
#pragma unroll
    for (int mt = 0; mt < 2; ++mt)
#pragma unroll
      for (int nt = 0; nt < 4; ++nt)
        acc[mt][nt] = wmma_bf16(af[mt].v, bfr[nt].v, acc[mt][nt]);
    __syncthreads();
  }

  // --- epilogue: scatter to Q / K / V^T (bf16) with bias add ---
#pragma unroll
  for (int nt = 0; nt < 4; ++nt) {
    int gn = n0 + wn * 64 + nt * 16 + l;
    float bv = bias[gn];
    int which = gn >> 10;          // 0=Q 1=K 2=V
    int d = gn & (DD - 1);
    int h = d >> 6, hd = d & (HDD - 1);
#pragma unroll
    for (int mt = 0; mt < 2; ++mt) {
#pragma unroll
      for (int v = 0; v < 8; ++v) {
        int gm = m0 + wm * 32 + mt * 16 + g * 8 + v;
        int b = gm >> 11, s = gm & (SS - 1);
        unsigned short o = f2bf(acc[mt][nt][v] + bv);
        size_t bh = (size_t)(b * HH + h);
        if (which == 0)      Qo[(bh * SS + s) * HDD + hd] = o;
        else if (which == 1) Ko[(bh * SS + s) * HDD + hd] = o;
        else                 Vt[(bh * HDD + hd) * SS + s] = o;
      }
    }
  }
}

// ---------------------------------------------------------------------------
// Kernel 2: flash attention.  Block = (b, h, 64-row Q tile), 4 waves x 16 rows.
// Streams 64-row KV tiles; online softmax; never materializes scores.
// ---------------------------------------------------------------------------
__global__ __launch_bounds__(128) void flash_kernel(
    const unsigned short* __restrict__ Q, const unsigned short* __restrict__ K,
    const unsigned short* __restrict__ Vt, const float* __restrict__ mask,
    unsigned short* __restrict__ Ctx) {
  __shared__ __align__(16) unsigned short Ps[4][16 * 80];  // per-wave P tile

  const int tid = threadIdx.x, lane = tid & 31, w = tid >> 5;
  const int g = lane >> 4, l = lane & 15;
  const int b = blockIdx.z, h = blockIdx.y, qt = blockIdx.x;
  const size_t bh = (size_t)(b * HH + h);
  const unsigned short* Qb = Q + bh * SS * HDD;
  const unsigned short* Kb = K + bh * SS * HDD;
  const unsigned short* Vb = Vt + bh * HDD * SS;

  // Q fragments (16 rows x 64 k), resident for whole kernel
  FragU qf[2];
  {
    int qrow = qt * 64 + w * 16 + l;
#pragma unroll
    for (int kc = 0; kc < 2; ++kc) {
      const unsigned short* p = Qb + (size_t)qrow * HDD + kc * 32;
      qf[kc].q[0] = *(const u32x4*)(p + g * 8);
      qf[kc].q[1] = *(const u32x4*)(p + 16 + g * 8);
    }
  }

  float mi[8], li[8];
#pragma unroll
  for (int v = 0; v < 8; ++v) { mi[v] = -1e30f; li[v] = 0.f; }
  f32x8 o[4];
#pragma unroll
  for (int c = 0; c < 4; ++c) o[c] = zero8();

  const float inv = 0.125f;  // 1/sqrt(64)

  for (int kt = 0; kt < SS / 64; ++kt) {
    // prefetch next K tile rows into cache
    if (kt + 1 < SS / 64)
      __builtin_prefetch(Kb + (size_t)((kt + 1) * 64 + l) * HDD, 0, 0);

    // K fragments: column n = kv row (HD-contiguous in memory)
    FragU kf[4][2];
#pragma unroll
    for (int nt = 0; nt < 4; ++nt)
#pragma unroll
      for (int kc = 0; kc < 2; ++kc) {
        const unsigned short* p =
            Kb + (size_t)(kt * 64 + nt * 16 + l) * HDD + kc * 32 + g * 8;
        kf[nt][kc].q[0] = *(const u32x4*)(p);
        kf[nt][kc].q[1] = *(const u32x4*)(p + 16);
      }

    // scores = Q K^T
    f32x8 sc[4];
#pragma unroll
    for (int nt = 0; nt < 4; ++nt) {
      sc[nt] = zero8();
      sc[nt] = wmma_bf16(qf[0].v, kf[nt][0].v, sc[nt]);
      sc[nt] = wmma_bf16(qf[1].v, kf[nt][1].v, sc[nt]);
    }

    // scale + additive mask (per kv column)
    float mk[4];
#pragma unroll
    for (int nt = 0; nt < 4; ++nt)
      mk[nt] = mask[b * SS + kt * 64 + nt * 16 + l];
#pragma unroll
    for (int nt = 0; nt < 4; ++nt)
#pragma unroll
      for (int v = 0; v < 8; ++v) sc[nt][v] = sc[nt][v] * inv + mk[nt];

    // online softmax per row (row = g*8+v across the 16-lane half)
#pragma unroll
    for (int v = 0; v < 8; ++v) {
      float mv = fmaxf(fmaxf(sc[0][v], sc[1][v]), fmaxf(sc[2][v], sc[3][v]));
      mv = halfmax(mv);
      float mn = fmaxf(mi[v], mv);
      float alpha = __expf(mi[v] - mn);
      float ps = 0.f;
#pragma unroll
      for (int nt = 0; nt < 4; ++nt) {
        float pv = __expf(sc[nt][v] - mn);
        sc[nt][v] = pv;
        ps += pv;
      }
      ps = halfsum(ps);
      li[v] = li[v] * alpha + ps;
      mi[v] = mn;
#pragma unroll
      for (int c = 0; c < 4; ++c) o[c][v] *= alpha;
    }

    // C-layout -> A-layout for P via per-wave LDS (no cross-wave sync needed)
    unsigned short* ps_ = Ps[w];
#pragma unroll
    for (int nt = 0; nt < 4; ++nt)
#pragma unroll
      for (int v = 0; v < 8; ++v)
        ps_[(g * 8 + v) * 80 + nt * 16 + l] = f2bf(sc[nt][v]);

    FragU pf[2];
#pragma unroll
    for (int kc = 0; kc < 2; ++kc) {
      pf[kc].q[0] = *(const u32x4*)(&ps_[l * 80 + kc * 32 + g * 8]);
      pf[kc].q[1] = *(const u32x4*)(&ps_[l * 80 + kc * 32 + 16 + g * 8]);
    }

    // O += P V  (V pre-transposed: column hd is S-contiguous)
#pragma unroll
    for (int c = 0; c < 4; ++c) {
#pragma unroll
      for (int kc = 0; kc < 2; ++kc) {
        FragU vf;
        const unsigned short* p =
            Vb + (size_t)(c * 16 + l) * SS + kt * 64 + kc * 32 + g * 8;
        vf.q[0] = *(const u32x4*)(p);
        vf.q[1] = *(const u32x4*)(p + 16);
        o[c] = wmma_bf16(pf[kc].v, vf.v, o[c]);
      }
    }
  }

  // normalize and write context (head-merged [B,S,D] bf16)
#pragma unroll
  for (int c = 0; c < 4; ++c) {
    int hd = c * 16 + l;
#pragma unroll
    for (int v = 0; v < 8; ++v) {
      int srow = qt * 64 + w * 16 + g * 8 + v;
      float val = o[c][v] / li[v];
      Ctx[((size_t)(b * SS + srow)) * DD + h * HDD + hd] = f2bf(val);
    }
  }
}

// ---------------------------------------------------------------------------
// Kernel 3: output projection.  ctx[4096,1024](bf16) @ w_proj[1024,1024] + b.
// A tile is already bf16 in the workspace -> DMA it straight into LDS with
// GLOBAL_LOAD_ASYNC_TO_LDS_B128 (ASYNCcnt), no VGPR round-trip.
// ---------------------------------------------------------------------------
__global__ __launch_bounds__(256) void proj_kernel(
    const unsigned short* __restrict__ Ctx, const float* __restrict__ W,
    const float* __restrict__ bias, float* __restrict__ Out) {
  __shared__ __align__(16) unsigned short As[128 * 48];
  __shared__ __align__(16) unsigned short Bs[128 * 48];

  const int tid = threadIdx.x;
  const int lane = tid & 31, wave = tid >> 5;
  const int wm = wave & 3, wn = wave >> 2;
  const int g = lane >> 4, l = lane & 15;
  const int m0 = blockIdx.y * 128, n0 = blockIdx.x * 128;

  f32x8 acc[2][4];
#pragma unroll
  for (int mt = 0; mt < 2; ++mt)
#pragma unroll
    for (int nt = 0; nt < 4; ++nt) acc[mt][nt] = zero8();

  for (int k0 = 0; k0 < DD; k0 += 32) {
    // A tile 128x32 bf16 = 512 16-byte slots: async DMA global -> LDS
#pragma unroll
    for (int i = 0; i < 2; ++i) {
      int slot = tid + i * 256;
      int r = slot >> 2, c8 = slot & 3;
      unsigned lds_addr = (unsigned)(size_t)&As[r * 48 + c8 * 8];
      unsigned long long gaddr =
          (unsigned long long)(size_t)(Ctx + (size_t)(m0 + r) * DD + k0 + c8 * 8);
      asm volatile("global_load_async_to_lds_b128 %0, %1, off"
                   :: "v"(lds_addr), "v"(gaddr)
                   : "memory");
    }
    // B 32x128 fp32 -> bf16 transposed (needs conversion: VGPR path)
#pragma unroll
    for (int i = 0; i < 4; ++i) {
      int slot = tid + i * 256;
      int kr = slot >> 5, c4 = slot & 31;
      f32x4 w = *(const f32x4*)(W + (size_t)(k0 + kr) * DD + n0 + c4 * 4);
#pragma unroll
      for (int j = 0; j < 4; ++j) Bs[(c4 * 4 + j) * 48 + kr] = f2bf(w[j]);
    }
    asm volatile("s_wait_asynccnt 0" ::: "memory");  // DMA'd A tile visible
    __syncthreads();   // also emits s_wait_dscnt 0 for the B-tile ds stores

    FragU af[2], bfr[4];
#pragma unroll
    for (int mt = 0; mt < 2; ++mt) {
      int row = wm * 32 + mt * 16 + l;
      af[mt].q[0] = *(const u32x4*)(&As[row * 48 + g * 8]);
      af[mt].q[1] = *(const u32x4*)(&As[row * 48 + 16 + g * 8]);
    }
#pragma unroll
    for (int nt = 0; nt < 4; ++nt) {
      int col = wn * 64 + nt * 16 + l;
      bfr[nt].q[0] = *(const u32x4*)(&Bs[col * 48 + g * 8]);
      bfr[nt].q[1] = *(const u32x4*)(&Bs[col * 48 + 16 + g * 8]);
    }
#pragma unroll
    for (int mt = 0; mt < 2; ++mt)
#pragma unroll
      for (int nt = 0; nt < 4; ++nt)
        acc[mt][nt] = wmma_bf16(af[mt].v, bfr[nt].v, acc[mt][nt]);
    __syncthreads();
  }

#pragma unroll
  for (int nt = 0; nt < 4; ++nt) {
    int gn = n0 + wn * 64 + nt * 16 + l;
    float bv = bias[gn];
#pragma unroll
    for (int mt = 0; mt < 2; ++mt)
#pragma unroll
      for (int v = 0; v < 8; ++v) {
        int gm = m0 + wm * 32 + mt * 16 + g * 8 + v;
        Out[(size_t)gm * DD + gn] = acc[mt][nt][v] + bv;
      }
  }
}

// ---------------------------------------------------------------------------
extern "C" void kernel_launch(void* const* d_in, const int* in_sizes, int n_in,
                              void* d_out, int out_size, void* d_ws,
                              size_t ws_size, hipStream_t stream) {
  const float* x      = (const float*)d_in[0];
  const float* amask  = (const float*)d_in[1];
  const float* w_attn = (const float*)d_in[2];
  const float* b_attn = (const float*)d_in[3];
  const float* w_proj = (const float*)d_in[4];
  const float* b_proj = (const float*)d_in[5];
  float* out = (float*)d_out;

  // workspace layout (bf16): Q | K | V^T | Ctx, each B*H*S*HD = 4,194,304 elems
  const size_t seg = (size_t)BB * HH * SS * HDD;  // elements
  unsigned short* Q   = (unsigned short*)d_ws;
  unsigned short* K   = Q + seg;
  unsigned short* Vt  = K + seg;
  unsigned short* Ctx = Vt + seg;

  qkv_kernel<<<dim3(N3 / 128, MM / 128), 256, 0, stream>>>(x, w_attn, b_attn,
                                                           Q, K, Vt);
  flash_kernel<<<dim3(SS / 64, HH, BB), 128, 0, stream>>>(Q, K, Vt, amask, Ctx);
  proj_kernel<<<dim3(DD / 128, MM / 128), 256, 0, stream>>>(Ctx, w_proj, b_proj,
                                                            out);
}
